// VQVAE_41970420418250
// MI455X (gfx1250) — matrix operand order repaired and checked by reading, not compiled, not run on hardware
//
#include <hip/hip_runtime.h>
#include <cmath>

// ---------------------------------------------------------------------------
// VQ-VAE forward for gfx1250 (MI455X). All conv / attention matmuls run
// through v_wmma_f32_16x16x32_f16 (f32 -> f16 operands, f32 accumulate).
// v3: templated kernel-shape (constant-divisor im2col), packed-f16 weights,
// double-buffered LDS staging (1 barrier / K-chunk), fused residual epilogue.
// ---------------------------------------------------------------------------

typedef _Float16 half_t;
typedef __attribute__((ext_vector_type(8)))  _Float16 v8h;
typedef __attribute__((ext_vector_type(16))) _Float16 v16h;
typedef __attribute__((ext_vector_type(8)))  float    v8f;

#define KC   32    // WMMA K chunk
#define TM   32    // M tile per block (2 wmma rows)
#define TNB  128   // N tile per block (8 wmma cols, 2 per wave)
#define LDP  40    // LDS row stride in halves (80 B = 5x16 B, keeps 16B align)

// Build a 16-wide fragment from two contiguous 8-wide LDS loads.
__device__ __forceinline__ v16h frag16(const half_t* p0, const half_t* p1)
{
    v8h lo = *(const v8h*)p0;
    v8h hi = *(const v8h*)p1;
    return __builtin_shufflevector(lo, hi, 0, 1, 2, 3, 4, 5, 6, 7,
                                   8, 9, 10, 11, 12, 13, 14, 15);
}

// ---------------------------------------------------------------------------
// Weight packer: f32 -> f16, layout [Cout][Cin*KH*KW] row-major.
// mode 1 folds the ConvTranspose2d (Cin,Cout,4,4) flip/transpose at pack time.
// ---------------------------------------------------------------------------
__global__ void __launch_bounds__(256)
k_pack_w(const float* __restrict__ w, half_t* __restrict__ o,
         int Cin, int Cout, int KH, int KW, int mode)
{
    int Ktot = Cin * KH * KW;
    int n = Cout * Ktot;
    for (int i = blockIdx.x * 256 + threadIdx.x; i < n; i += gridDim.x * 256) {
        float v;
        if (mode == 0) {
            v = w[i];                               // OIHW is already [Cout][Ktot]
        } else {
            int m = i / Ktot, k = i % Ktot;
            int KK = KH * KW, ci = k / KK, r = k % KK, ky = r / KW, kx = r % KW;
            v = w[(((size_t)ci * Cout + m) * KH + (KH - 1 - ky)) * KW + (KW - 1 - kx)];
        }
        o[i] = (half_t)v;
    }
}

// ---------------------------------------------------------------------------
// Implicit-GEMM convolution, weights pre-packed f16 [Cout][Ktot].
//   MODE 0: normal conv (stride/pad); MODE 1: ConvTranspose2d(k=4,s=2,p=1).
// M = Cout, N = OH*OW, K = Cin*KHt*KWt. 256 threads = 8 waves; wave w ->
// (mt = w&1, nt = w>>1); two 16x16 subtiles per wave sharing one A fragment.
// Double-buffered LDS: stage chunk c+1 while WMMAs run on chunk c.
// Optional fused epilogue: y = acc + bias + addend.
// ---------------------------------------------------------------------------
template <int KHt, int KWt, int MODE>
__global__ void __launch_bounds__(256)
k_conv_wmma(const float* __restrict__ x, const half_t* __restrict__ w16,
            const float* __restrict__ bias, const float* __restrict__ addend,
            float* __restrict__ y,
            int Cin, int Cout, int IH, int IW, int OH, int OW,
            int stride, int pad)
{
    __shared__ half_t As[2][TM][LDP];    // A tile, row-major (M x K)
    __shared__ half_t Bs[2][TNB][LDP];   // B tile, N-major   (N x K)

    const int b    = blockIdx.z;
    const int m0   = blockIdx.y * TM;
    const int n0   = blockIdx.x * TNB;
    const int tid  = threadIdx.x;
    const int lane = tid & 31;
    const int wv   = tid >> 5;
    const int mt   = wv & 1;
    const int nt   = wv >> 1;
    const int KK   = KHt * KWt;
    const int Ktot = Cin * KK;
    const int Npix = OH * OW;
    const float* xb = x + (size_t)b * Cin * IH * IW;

    auto stage = [&](int k0, int sb) {
        // A: TM x KC packed-f16 weights (branchless copy)
        for (int i = tid; i < TM * KC; i += 256) {
            int m = i / KC, kk = i % KC, k = k0 + kk;
            half_t v = (half_t)0.f;
            if ((m0 + m) < Cout && k < Ktot)
                v = w16[(size_t)(m0 + m) * Ktot + k];
            As[sb][m][kk] = v;
        }
        // B: KC x TNB im2col patch, stored N-major (f32 -> f16)
        for (int i = tid; i < KC * TNB; i += 256) {
            int n = i % TNB, kk = i / TNB;
            int k = k0 + kk, ng = n0 + n;
            float v = 0.f;
            if (k < Ktot && ng < Npix) {
                int ci = k / KK, r = k % KK, ky = r / KWt, kx = r % KWt;
                int oy = ng / OW, ox = ng % OW;
                if (MODE == 0) {
                    int iy = oy * stride - pad + ky;
                    int ix = ox * stride - pad + kx;
                    if (iy >= 0 && iy < IH && ix >= 0 && ix < IW)
                        v = xb[((size_t)ci * IH + iy) * IW + ix];
                } else {
                    int ty = oy + 1 - ky, tx = ox + 1 - kx;
                    if (ty >= 0 && tx >= 0 && !(ty & 1) && !(tx & 1)) {
                        int iy = ty >> 1, ix = tx >> 1;
                        if (iy < IH && ix < IW)
                            v = xb[((size_t)ci * IH + iy) * IW + ix];
                    }
                }
            }
            Bs[sb][n][kk] = (half_t)v;
        }
    };

    v8f acc0 = {0.f, 0.f, 0.f, 0.f, 0.f, 0.f, 0.f, 0.f};
    v8f acc1 = {0.f, 0.f, 0.f, 0.f, 0.f, 0.f, 0.f, 0.f};

    const int nch = (Ktot + KC - 1) / KC;
    stage(0, 0);
    __syncthreads();

    for (int c = 0; c < nch; ++c) {
        const int cur = c & 1;
        if (c + 1 < nch) {
            // prefetch chunk c+2 weights; stage chunk c+1 into alternate buffer
            if (tid < TM && (c + 2) * KC < Ktot && (m0 + tid) < Cout)
                __builtin_prefetch(&w16[(size_t)(m0 + tid) * Ktot + (c + 2) * KC], 0, 3);
            stage((c + 1) * KC, cur ^ 1);
        }

        // fragments per ISA 7.12.2 wave32 layouts; contiguous 16B LDS reads
        const int mr  = mt * 16 + (lane & 15);
        const int kb  = (lane >> 4) * 8;    // A: lanes 0-15 K0-7/16-23; 16-31 K8-15/24-31
        const int kb2 = (lane >> 4) * 16;   // B: same K order as A (via kb)
        const int nc  = nt * 32 + (lane & 15);

        v16h af  = frag16(&As[cur][mr][kb], &As[cur][mr][16 + kb]);
        v16h bf0 = frag16(&Bs[cur][nc][kb], &Bs[cur][nc][16 + kb]);
        v16h bf1 = frag16(&Bs[cur][nc + 16][kb], &Bs[cur][nc + 16][16 + kb]);

        acc0 = __builtin_amdgcn_wmma_f32_16x16x32_f16(false, af, false, bf0,
                                                      (short)0, acc0, false, false);
        acc1 = __builtin_amdgcn_wmma_f32_16x16x32_f16(false, af, false, bf1,
                                                      (short)0, acc1, false, false);
        __syncthreads();    // buffer cur may be restaged next iteration
    }

    // C/D layout: VGPR r -> M = r (lanes 0-15) / 8+r (lanes 16-31); N = lane&15
    const int mb = m0 + mt * 16 + (lane >> 4) * 8;
    const int nc = n0 + nt * 32 + (lane & 15);
#pragma unroll
    for (int r = 0; r < 8; ++r) {
        int mg = mb + r;
        if (mg < Cout) {
            float bv = bias ? bias[mg] : 0.f;
            size_t rowb = ((size_t)b * Cout + mg) * Npix;
            if (nc < Npix) {
                float v = acc0[r] + bv;
                if (addend) v += addend[rowb + nc];
                y[rowb + nc] = v;
            }
            if (nc + 16 < Npix) {
                float v = acc1[r] + bv;
                if (addend) v += addend[rowb + nc + 16];
                y[rowb + nc + 16] = v;
            }
        }
    }
}

// ---------------------------------------------------------------------------
// Batched-strided WMMA GEMM: C = alpha * A x B (f32 in, f16 MACs, f32 acc).
// TA: A[k,m] (else A[m,k]); TB: B[n,k] (else B[k,n]).
// Per-z offsets: off = (z / innerCnt)*sXo + (z % innerCnt)*sXi.
// ---------------------------------------------------------------------------
template <int TA, int TB>
__global__ void __launch_bounds__(256)
k_gemm_wmma(const float* __restrict__ A, const float* __restrict__ B,
            float* __restrict__ Cm,
            int M, int N, int Kt, int lda, int ldb, int ldc,
            float alpha, int innerCnt,
            long long sAo, long long sAi, long long sBo, long long sBi,
            long long sCo, long long sCi)
{
    __shared__ half_t As[2][TM][LDP];
    __shared__ half_t Bs[2][TNB][LDP];

    const int bo = blockIdx.z / innerCnt, bi = blockIdx.z % innerCnt;
    A  += (size_t)bo * sAo + (size_t)bi * sAi;
    B  += (size_t)bo * sBo + (size_t)bi * sBi;
    Cm += (size_t)bo * sCo + (size_t)bi * sCi;

    const int m0   = blockIdx.y * TM;
    const int n0   = blockIdx.x * TNB;
    const int tid  = threadIdx.x;
    const int lane = tid & 31;
    const int wv   = tid >> 5;
    const int mt   = wv & 1;
    const int nt   = wv >> 1;

    auto stage = [&](int k0, int sb) {
        for (int i = tid; i < TM * KC; i += 256) {
            int m = i / KC, kk = i % KC, k = k0 + kk, mg = m0 + m;
            float v = 0.f;
            if (mg < M && k < Kt)
                v = TA ? A[(size_t)k * lda + mg] : A[(size_t)mg * lda + k];
            As[sb][m][kk] = (half_t)v;
        }
        for (int i = tid; i < KC * TNB; i += 256) {
            int n = i % TNB, kk = i / TNB, k = k0 + kk, ng = n0 + n;
            float v = 0.f;
            if (k < Kt && ng < N)
                v = TB ? B[(size_t)ng * ldb + k] : B[(size_t)k * ldb + ng];
            Bs[sb][n][kk] = (half_t)v;
        }
    };

    v8f acc0 = {0.f, 0.f, 0.f, 0.f, 0.f, 0.f, 0.f, 0.f};
    v8f acc1 = {0.f, 0.f, 0.f, 0.f, 0.f, 0.f, 0.f, 0.f};

    const int nch = (Kt + KC - 1) / KC;
    stage(0, 0);
    __syncthreads();

    for (int c = 0; c < nch; ++c) {
        const int cur = c & 1;
        if (c + 1 < nch) stage((c + 1) * KC, cur ^ 1);

        const int mr  = mt * 16 + (lane & 15);
        const int kb  = (lane >> 4) * 8;
        const int kb2 = (lane >> 4) * 16;
        const int nc  = nt * 32 + (lane & 15);

        v16h af  = frag16(&As[cur][mr][kb], &As[cur][mr][16 + kb]);
        v16h bf0 = frag16(&Bs[cur][nc][kb], &Bs[cur][nc][16 + kb]);
        v16h bf1 = frag16(&Bs[cur][nc + 16][kb], &Bs[cur][nc + 16][16 + kb]);

        acc0 = __builtin_amdgcn_wmma_f32_16x16x32_f16(false, af, false, bf0,
                                                      (short)0, acc0, false, false);
        acc1 = __builtin_amdgcn_wmma_f32_16x16x32_f16(false, af, false, bf1,
                                                      (short)0, acc1, false, false);
        __syncthreads();
    }

    const int mb = m0 + mt * 16 + (lane >> 4) * 8;
    const int nc = n0 + nt * 32 + (lane & 15);
#pragma unroll
    for (int r = 0; r < 8; ++r) {
        int mg = mb + r;
        if (mg < M) {
            if (nc < N)      Cm[(size_t)mg * ldc + nc]      = alpha * acc0[r];
            if (nc + 16 < N) Cm[(size_t)mg * ldc + nc + 16] = alpha * acc1[r];
        }
    }
}

// ---------------------------------------------------------------------------
// GroupNorm (+ optional SiLU). One block per (batch, group).
// ---------------------------------------------------------------------------
__global__ void __launch_bounds__(256)
k_groupnorm_act(const float* __restrict__ x, const float* __restrict__ g,
                const float* __restrict__ bta, float* __restrict__ y,
                int C, int Gr, int HW, int silu)
{
    __shared__ float s1[256], s2[256];
    int b = blockIdx.x / Gr, grp = blockIdx.x % Gr;
    int cpg = C / Gr;
    size_t base = ((size_t)b * C + (size_t)grp * cpg) * HW;
    int n = cpg * HW;
    int t = threadIdx.x;
    float sum = 0.f, sq = 0.f;
    for (int i = t; i < n; i += 256) { float v = x[base + i]; sum += v; sq += v * v; }
    s1[t] = sum; s2[t] = sq; __syncthreads();
    for (int o = 128; o > 0; o >>= 1) {
        if (t < o) { s1[t] += s1[t + o]; s2[t] += s2[t + o]; }
        __syncthreads();
    }
    float mu   = s1[0] / n;
    float var  = s2[0] / n - mu * mu;
    float rstd = rsqrtf(var + 1e-5f);
    for (int i = t; i < n; i += 256) {
        int c = grp * cpg + i / HW;
        float v = (x[base + i] - mu) * rstd * g[c] + bta[c];
        if (silu) v = v / (1.f + __expf(-v));
        y[base + i] = v;
    }
}

__global__ void __launch_bounds__(256)
k_softmax_rows(float* __restrict__ s, int L)
{
    __shared__ float red[256];
    size_t row = (size_t)blockIdx.x * L;
    int t = threadIdx.x;
    float m = -3.4e38f;
    for (int i = t; i < L; i += 256) m = fmaxf(m, s[row + i]);
    red[t] = m; __syncthreads();
    for (int o = 128; o > 0; o >>= 1) {
        if (t < o) red[t] = fmaxf(red[t], red[t + o]);
        __syncthreads();
    }
    m = red[0]; __syncthreads();
    float sum = 0.f;
    for (int i = t; i < L; i += 256) { float e = __expf(s[row + i] - m); s[row + i] = e; sum += e; }
    red[t] = sum; __syncthreads();
    for (int o = 128; o > 0; o >>= 1) {
        if (t < o) red[t] += red[t + o];
        __syncthreads();
    }
    float inv = 1.f / red[0];
    for (int i = t; i < L; i += 256) s[row + i] *= inv;
}

__global__ void k_zero1(float* p) { if (threadIdx.x == 0 && blockIdx.x == 0) p[0] = 0.f; }

// Codebook argmin quantization + squared-error accumulation.
__global__ void __launch_bounds__(256)
k_quantize(const float* __restrict__ z, const float* __restrict__ emb,
           float* __restrict__ qbuf, float* __restrict__ qout,
           float* __restrict__ acc, int Bn, int C, int HW, int K)
{
    int n = blockIdx.x * 256 + threadIdx.x;
    if (n >= Bn * HW) return;
    int b = n / HW, pix = n % HW;
    float zf[8];
    for (int c = 0; c < C && c < 8; ++c) zf[c] = z[((size_t)b * C + c) * HW + pix];
    int best = 0; float bd = 3.4e38f;
    for (int kk = 0; kk < K; ++kk) {
        float d = 0.f;
        for (int c = 0; c < C; ++c) { float t = zf[c] - emb[(size_t)kk * C + c]; d += t * t; }
        if (d < bd) { bd = d; best = kk; }
    }
    float err = 0.f;
    for (int c = 0; c < C; ++c) {
        float q = emb[(size_t)best * C + c];
        float t = q - zf[c]; err += t * t;
        qbuf[((size_t)b * C + c) * HW + pix] = q;
        qout[((size_t)b * C + c) * HW + pix] = q;
    }
    atomicAdd(acc, err);
}

// cb_loss == commit_loss numerically (stop_gradient is identity in value).
__global__ void k_finalize(const float* acc, float* losses, float inv_n)
{
    if (threadIdx.x == 0 && blockIdx.x == 0) {
        float v = acc[0] * inv_n;
        losses[0] = v;   // cb_loss
        losses[1] = v;   // commit_loss
    }
}

// ---------------------------------------------------------------------------
// Host-side helpers
// ---------------------------------------------------------------------------
static void conv_l(hipStream_t st, const float* x, const float* w, const float* b,
                   const float* addend, float* y, half_t*& wpool,
                   int Bn, int Cin, int Cout, int IH, int IW,
                   int KH, int KW, int stride, int pad, int mode)
{
    int OH, OW;
    if (mode) { OH = IH * 2; OW = IW * 2; }
    else { OH = (IH + 2 * pad - KH) / stride + 1; OW = (IW + 2 * pad - KW) / stride + 1; }

    // pack weights f32 -> f16 [Cout][Ktot] (deterministic bump allocation)
    int nels = Cout * Cin * KH * KW;
    half_t* wp = wpool;
    wpool += (size_t)((nels + 63) & ~63);
    int pg = (nels + 255) / 256; if (pg > 1024) pg = 1024;
    k_pack_w<<<dim3((unsigned)pg), 256, 0, st>>>(w, wp, Cin, Cout, KH, KW, mode);

    dim3 g((unsigned)((OH * OW + TNB - 1) / TNB), (unsigned)((Cout + TM - 1) / TM),
           (unsigned)Bn);
    if (mode == 1)
        k_conv_wmma<4, 4, 1><<<g, 256, 0, st>>>(x, wp, b, addend, y, Cin, Cout,
                                                IH, IW, OH, OW, stride, pad);
    else if (KH == 3)
        k_conv_wmma<3, 3, 0><<<g, 256, 0, st>>>(x, wp, b, addend, y, Cin, Cout,
                                                IH, IW, OH, OW, stride, pad);
    else if (KH == 4)
        k_conv_wmma<4, 4, 0><<<g, 256, 0, st>>>(x, wp, b, addend, y, Cin, Cout,
                                                IH, IW, OH, OW, stride, pad);
    else
        k_conv_wmma<1, 1, 0><<<g, 256, 0, st>>>(x, wp, b, addend, y, Cin, Cout,
                                                IH, IW, OH, OW, stride, pad);
}

static void gn_l(hipStream_t st, const float* x, const float* g, const float* b,
                 float* y, int Bn, int C, int G, int HW, int silu)
{
    k_groupnorm_act<<<dim3((unsigned)(Bn * G)), 256, 0, st>>>(x, g, b, y, C, G, HW, silu);
}

// Res-block param leaves are 10 consecutive entries:
// [c1.b, c1.w, c2.b, c2.w, gn1.b, gn1.g, gn2.b, gn2.g, res.b, res.w]
// Schedule: gn1(src->t1); c1(t1->t2); gn2(t2->t1); shortcut(src->t2);
//           c2(t1, addend=t2 -> dst)   (residual add fused into conv epilogue)
static void resnet_l(hipStream_t st, const float* const* P, int base, half_t*& wpool,
                     const float* src, float* dst, float* t1, float* t2,
                     int Cin, int Cout, int H, int W, int Bn, int G)
{
    gn_l(st, src, P[base + 5], P[base + 4], t1, Bn, Cin, G, H * W, 1);
    conv_l(st, t1, P[base + 1], P[base + 0], nullptr, t2, wpool,
           Bn, Cin, Cout, H, W, 3, 3, 1, 1, 0);
    gn_l(st, t2, P[base + 7], P[base + 6], t1, Bn, Cout, G, H * W, 1);
    conv_l(st, src, P[base + 9], P[base + 8], nullptr, t2, wpool,
           Bn, Cin, Cout, H, W, 1, 1, 1, 0, 0);
    conv_l(st, t1, P[base + 3], P[base + 2], /*addend=*/t2, dst, wpool,
           Bn, Cout, Cout, H, W, 3, 3, 1, 1, 0);
}

// Attn param leaves: [gn.b, gn.g, in_w, out_w] at base..base+3. x is (B,C,L), in-place.
static void attn_l(hipStream_t st, const float* const* P, int base, half_t*& wpool,
                   float* x, float* t0, float* qkv, float* scr, float* obuf,
                   int Bn, int C, int L, int G, int heads)
{
    gn_l(st, x, P[base + 1], P[base + 0], t0, Bn, C, G, L, 0);
    conv_l(st, t0, P[base + 2], nullptr, nullptr, qkv, wpool,
           Bn, C, 3 * C, 1, L, 1, 1, 1, 0, 0);
    int dh = C / heads;
    // scores[b,h,i,j] = (1/sqrt(dh)) * sum_d q[d,i] k[d,j]   (K = dh = 32: one chunk)
    {
        dim3 g((unsigned)((L + TNB - 1) / TNB), (unsigned)((L + TM - 1) / TM),
               (unsigned)(Bn * heads));
        k_gemm_wmma<1, 0><<<g, 256, 0, st>>>(qkv, qkv + (size_t)C * L, scr,
            L, L, dh, L, L, L, 1.0f / sqrtf((float)dh), heads,
            (long long)3 * C * L, (long long)dh * L,
            (long long)3 * C * L, (long long)dh * L,
            (long long)heads * L * L, (long long)L * L);
    }
    k_softmax_rows<<<dim3((unsigned)(Bn * heads * L)), 256, 0, st>>>(scr, L);
    // o[d,i] = sum_j v[d,j] * att[i,j]   (output laid out (B,C,L))
    {
        dim3 g((unsigned)((L + TNB - 1) / TNB), (unsigned)((dh + TM - 1) / TM),
               (unsigned)(Bn * heads));
        k_gemm_wmma<0, 1><<<g, 256, 0, st>>>(qkv + (size_t)2 * C * L, scr, obuf,
            dh, L, L, L, L, L, 1.0f, heads,
            (long long)3 * C * L, (long long)dh * L,
            (long long)heads * L * L, (long long)L * L,
            (long long)C * L, (long long)dh * L);
    }
    // out-projection with fused residual add: x = x + out_w @ o
    conv_l(st, obuf, P[base + 3], nullptr, /*addend=*/x, x, wpool,
           Bn, C, C, 1, L, 1, 1, 1, 0, 0);
}

static void mid_l(hipStream_t st, const float* const* P, int attn_base, int res0, int res1,
                  half_t*& wpool, float* x, float* t0, float* t2, float* qkv, float* scr,
                  int Bn, int C, int H, int W, int G, int heads)
{
    resnet_l(st, P, res0, wpool, x, x, t0, t2, C, C, H, W, Bn, G);
    attn_l(st, P, attn_base, wpool, x, t0, qkv, scr, t2, Bn, C, H * W, G, heads);
    resnet_l(st, P, res1, wpool, x, x, t0, t2, C, C, H, W, Bn, G);
}

// ---------------------------------------------------------------------------
// d_in layout assumption: [0] = x, then params flattened pytree-style
// (dict keys sorted, lists in order). Leaf index map (conv={b,w}, gn={b,g}):
//   1 codebook | 2-3 dec_in{b,w}
//   dec_mid[0]: 4-7 attn{gn.b,gn.g,in_w,out_w}; 8-17 res0; 18-27 res1
//   28-29 dec_norm{b,g} | 30-31 dec_out{b,w}
//   down0: 32-33 ds{b,w}, 34-43 res | down1: 44-45 ds, 46-55 res | down2: 56-57 ds, 58-67 res
//   68-69 enc_in{b,w}
//   enc_mid[0]: 70-73 attn; 74-83 res0; 84-93 res1
//   94-95 enc_norm | 96-97 enc_out | 98-99 post_q | 100-101 pre_q
//   up0: 102-111 res, 112-113 us{b,w} | up1: 114-123 res, 124-125 us | up2: 126-135 res, 136-137 us
// ---------------------------------------------------------------------------
extern "C" void kernel_launch(void* const* d_in, const int* in_sizes, int n_in,
                              void* d_out, int out_size, void* d_ws, size_t ws_size,
                              hipStream_t stream)
{
    (void)in_sizes; (void)n_in; (void)out_size; (void)ws_size;
    const float* P[160];
    for (int i = 0; i < n_in && i < 160; ++i) P[i] = (const float*)d_in[i];
    float* ws  = (float*)d_ws;
    float* out = (float*)d_out;

    const int Bn = 4, G = 32, HEADS = 8;
    const long long BIG = 8388608LL;        // (4,128,128,128) floats
    float* b0  = ws;
    float* b1  = ws + BIG;
    float* b2  = ws + 2 * BIG;
    float* qkv = ws + 3 * BIG;              // (4,768,256) = 786432
    float* scr = qkv + 786432;              // (32,256,256) = 2097152
    float* acc = scr + 2097152;             // 1 float (loss accumulator)
    half_t* wpool = (half_t*)(acc + 64);    // packed-f16 weight pool (~29 MB)
    half_t* wp = wpool;                     // bump cursor, reset every launch

    // ---- encoder ----
    conv_l(stream, P[0], P[69], P[68], nullptr, b0, wp, Bn, 3, 64, 128, 128,
           3, 3, 1, 1, 0);                                                    // enc_in
    resnet_l(stream, P, 34, wp, b0, b0, b1, b2, 64, 128, 128, 128, Bn, G);    // down0.res
    conv_l(stream, b0, P[33], P[32], nullptr, b1, wp, Bn, 128, 128, 128, 128,
           4, 4, 2, 1, 0);                                                    // down0.ds -> 64
    resnet_l(stream, P, 46, wp, b1, b1, b0, b2, 128, 256, 64, 64, Bn, G);     // down1.res
    conv_l(stream, b1, P[45], P[44], nullptr, b0, wp, Bn, 256, 256, 64, 64,
           4, 4, 2, 1, 0);                                                    // down1.ds -> 32
    resnet_l(stream, P, 58, wp, b0, b0, b1, b2, 256, 256, 32, 32, Bn, G);     // down2.res
    conv_l(stream, b0, P[57], P[56], nullptr, b1, wp, Bn, 256, 256, 32, 32,
           4, 4, 2, 1, 0);                                                    // down2.ds -> 16
    mid_l(stream, P, 70, 74, 84, wp, b1, b0, b2, qkv, scr,
          Bn, 256, 16, 16, G, HEADS);                                         // enc_mid
    gn_l(stream, b1, P[95], P[94], b0, Bn, 256, G, 256, 1);                   // enc_norm+silu
    conv_l(stream, b0, P[97], P[96], nullptr, b2, wp, Bn, 256, 4, 16, 16,
           3, 3, 1, 1, 0);                                                    // enc_out
    conv_l(stream, b2, P[101], P[100], nullptr, b0, wp, Bn, 4, 4, 16, 16,
           1, 1, 1, 0, 0);                                                    // pre_q

    // ---- quantize ----
    k_zero1<<<1, 1, 0, stream>>>(acc);
    k_quantize<<<dim3((unsigned)((Bn * 256 + 255) / 256)), 256, 0, stream>>>(
        b0, P[1], b2, out + 196608, acc, Bn, 4, 256, 8192);
    k_finalize<<<1, 1, 0, stream>>>(acc, out + 196608 + 4096, 1.0f / 4096.0f);

    // ---- decoder ----
    conv_l(stream, b2, P[99], P[98], nullptr, b0, wp, Bn, 4, 4, 16, 16,
           1, 1, 1, 0, 0);                                                    // post_q
    conv_l(stream, b0, P[3], P[2], nullptr, b1, wp, Bn, 4, 256, 16, 16,
           3, 3, 1, 1, 0);                                                    // dec_in
    mid_l(stream, P, 4, 8, 18, wp, b1, b0, b2, qkv, scr,
          Bn, 256, 16, 16, G, HEADS);                                         // dec_mid
    conv_l(stream, b1, P[113], P[112], nullptr, b0, wp, Bn, 256, 256, 16, 16,
           4, 4, 2, 1, 1);                                                    // up0.us -> 32
    resnet_l(stream, P, 102, wp, b0, b0, b1, b2, 256, 256, 32, 32, Bn, G);    // up0.res
    conv_l(stream, b0, P[125], P[124], nullptr, b1, wp, Bn, 256, 256, 32, 32,
           4, 4, 2, 1, 1);                                                    // up1.us -> 64
    resnet_l(stream, P, 114, wp, b1, b1, b0, b2, 256, 128, 64, 64, Bn, G);    // up1.res
    conv_l(stream, b1, P[137], P[136], nullptr, b0, wp, Bn, 128, 128, 64, 64,
           4, 4, 2, 1, 1);                                                    // up2.us -> 128
    resnet_l(stream, P, 126, wp, b0, b0, b1, b2, 128, 64, 128, 128, Bn, G);   // up2.res
    gn_l(stream, b0, P[29], P[28], b1, Bn, 64, G, 16384, 1);                  // dec_norm+silu
    conv_l(stream, b1, P[31], P[30], nullptr, out, wp, Bn, 64, 3, 128, 128,
           3, 3, 1, 1, 0);                                                    // dec_out
}
